// ClusteringJdcGCN_38388417691940
// MI455X (gfx1250) — matrix operand order repaired and checked by reading, not compiled, not run on hardware
//
#include <hip/hip_runtime.h>

// ---------------------------------------------------------------------------
// MI455X (gfx1250) multi-view GCN + attention + autoencoder.
// All GEMMs run through V_WMMA_F32_16X16X32_BF16 (wave32). B operands are
// kept transposed (BT[N x K]) so LDS staging and per-lane fragment builds are
// pure b128 traffic. Dims padded: M%128==0, K%32==0, N%128==0. Epilogues are
// template-specialized (no runtime branches), biases/outputs zero-padded so
// every load/store is guard-free; address math is strength-reduced.
// ---------------------------------------------------------------------------

typedef __attribute__((ext_vector_type(16))) __bf16 v16bf;
typedef __attribute__((ext_vector_type(8)))  __bf16 v8bf;
typedef __attribute__((ext_vector_type(8)))  float  v8f;

#define BM 128
#define BN 128
#define BK 32
#define LDT (BK + 8)   // LDS row stride in elements (80B, 16B-aligned)

static __device__ __forceinline__ unsigned short f2bf_bits(float f) {
  unsigned u = __builtin_bit_cast(unsigned, f);
  unsigned r = u + 0x7FFFu + ((u >> 16) & 1u);   // round-to-nearest-even
  return (unsigned short)(r >> 16);
}

// ---------------- conversion kernels ----------------------------------------

// f32 -> bf16, 8 elements per thread (two b128 loads, one b128 store). n8 = n/8.
extern "C" __global__ void __launch_bounds__(256)
k_cvt_bf16(const float* __restrict__ src, unsigned short* __restrict__ dst, long n8) {
  long i = (long)blockIdx.x * blockDim.x + threadIdx.x;
  long stride = (long)gridDim.x * blockDim.x;
  for (; i < n8; i += stride) {
    const float4* s = (const float4*)(src + i * 8);
    float4 x = s[0], y = s[1];
    union { unsigned short h[8]; uint4 q; } o;
    o.h[0] = f2bf_bits(x.x); o.h[1] = f2bf_bits(x.y);
    o.h[2] = f2bf_bits(x.z); o.h[3] = f2bf_bits(x.w);
    o.h[4] = f2bf_bits(y.x); o.h[5] = f2bf_bits(y.y);
    o.h[6] = f2bf_bits(y.z); o.h[7] = f2bf_bits(y.w);
    *(uint4*)(dst + i * 8) = o.q;
  }
}

// src [R x C] f32 row-major -> dst [C x R] bf16 (LDS-tiled transpose).
// R, C multiples of 32. Grid: (C/32, R/32), 256 threads.
extern "C" __global__ void __launch_bounds__(256)
k_cvt_t(const float* __restrict__ src, unsigned short* __restrict__ dst,
        int R, int C) {
  __shared__ float tile[32][33];
  const int c0 = blockIdx.x * 32, r0 = blockIdx.y * 32;
  const int tx = threadIdx.x & 31, ty = threadIdx.x >> 5;
#pragma unroll
  for (int i = ty; i < 32; i += 8)
    tile[i][tx] = src[(long)(r0 + i) * C + (c0 + tx)];
  __syncthreads();
#pragma unroll
  for (int i = ty; i < 32; i += 8)
    dst[(long)(c0 + i) * R + (r0 + tx)] = f2bf_bits(tile[tx][i]);
}

// weights: src [K x Ncol] f32 -> dst [Np x Kp] bf16 TRANSPOSED, zero-padded.
// Kp = 1<<kpShift (power of two).
extern "C" __global__ void __launch_bounds__(256)
k_cvt_pad_t(const float* __restrict__ src, int K, int Ncol,
            unsigned short* __restrict__ dst, int kpShift, int Np) {
  const int  kMask = (1 << kpShift) - 1;
  long total = (long)Np << kpShift;
  long i = (long)blockIdx.x * blockDim.x + threadIdx.x;
  long stride = (long)gridDim.x * blockDim.x;
  for (; i < total; i += stride) {
    int n = (int)(i >> kpShift), k = (int)i & kMask;
    float v = (k < K && n < Ncol) ? src[(long)k * Ncol + n] : 0.0f;
    dst[i] = f2bf_bits(v);
  }
}

// bias: src [n] f32 -> dst [np] f32 zero-padded
extern "C" __global__ void __launch_bounds__(256)
k_pad_bias(const float* __restrict__ src, int n, float* __restrict__ dst, int np) {
  int i = blockIdx.x * blockDim.x + threadIdx.x;
  if (i < np) dst[i] = (i < n) ? src[i] : 0.0f;
}

// z extraction: src [rows x 128] -> dst [rows x 10]; 16 lanes per row.
extern "C" __global__ void __launch_bounds__(256)
k_copy_z(const float* __restrict__ src, float* __restrict__ dst, long rows16) {
  long i = (long)blockIdx.x * blockDim.x + threadIdx.x;
  if (i >= rows16) return;
  long r = i >> 4;
  int  c = (int)i & 15;
  if (c < 10) dst[r * 10 + c] = src[(r << 7) + c];
}

// ---------------- bf16 WMMA GEMM (template-specialized epilogue) ------------
// C[M x N] = act(A[M x K] @ BT[N x K]^T + bias).  M%128==0, K%32==0, N%128==0.
// ACT: 0 none, 1 relu. bias (if HAS_BIAS) is zero-padded to N. Stores are
// guard-free: f32 row-major (HAS_CF), bf16 row-major or transposed (CBT).

template <int ACT, bool HAS_BIAS, bool HAS_CF, bool HAS_CB, bool CBT>
__global__ void __launch_bounds__(256)
k_gemm(const unsigned short* __restrict__ A, int lda,
       const unsigned short* __restrict__ BT, int ldb, int K,
       const float* __restrict__ bias,
       float* __restrict__ Cf, int ldcf,
       unsigned short* __restrict__ Cb, int ldcb) {
  __shared__ unsigned short sA[BM * LDT];
  __shared__ unsigned short sB[BN * LDT];

  const int tid  = threadIdx.x;
  const int lane = tid & 31;
  const int wave = tid >> 5;       // 0..7
  const int wm   = wave & 3;       // 4 row groups of 32
  const int wn   = wave >> 2;      // 2 col groups of 64
  const int l15  = lane & 15;
  const int hi   = lane >> 4;

  const long blockM = (long)blockIdx.y * BM;
  const long blockN = (long)blockIdx.x * BN;

  // guard-free b128 staging coordinates
  const int sr = tid >> 2;             // 0..63
  const int sc = (tid & 3) * 8;        // 0,8,16,24
  const unsigned short* aPtr = A  + (blockM + sr) * (long)lda + sc;
  const unsigned short* bPtr = BT + (blockN + sr) * (long)ldb + sc;

  const v8f vzero = {0.f, 0.f, 0.f, 0.f, 0.f, 0.f, 0.f, 0.f};
  v8f acc[2][4];
#pragma unroll
  for (int tr = 0; tr < 2; ++tr)
#pragma unroll
    for (int tc = 0; tc < 4; ++tc) acc[tr][tc] = vzero;

  for (int k0 = 0; k0 < K; k0 += BK) {
    uint4 a0 = *(const uint4*)(aPtr + k0);
    uint4 a1 = *(const uint4*)(aPtr + (long)64 * lda + k0);
    uint4 b0 = *(const uint4*)(bPtr + k0);
    uint4 b1 = *(const uint4*)(bPtr + (long)64 * ldb + k0);
    *(uint4*)&sA[sr * LDT + sc]        = a0;
    *(uint4*)&sA[(sr + 64) * LDT + sc] = a1;
    *(uint4*)&sB[sr * LDT + sc]        = b0;
    *(uint4*)&sB[(sr + 64) * LDT + sc] = b1;
    // prefetch next K tile into cache while we compute this one
    __builtin_prefetch(aPtr + k0 + BK, 0, 3);
    __builtin_prefetch(bPtr + k0 + BK, 0, 3);
    __syncthreads();

    // A fragment (16x32 bf16): elems 0..7 = K koff..koff+7, 8..15 = 16+koff..
    v16bf af[2];
#pragma unroll
    for (int tr = 0; tr < 2; ++tr) {
      const unsigned short* ap = &sA[(wm * 32 + tr * 16 + l15) * LDT];
      union { v16bf v; v8bf h[2]; } t;
      t.h[0] = *(const v8bf*)(ap + hi * 8);
      t.h[1] = *(const v8bf*)(ap + 16 + hi * 8);
      af[tr] = t.v;
    }
    // B fragment (32x16 bf16): lane = col, elems 0..7 = K hi*8.., 8..15 = 16+hi*8..
    v16bf bf[4];
#pragma unroll
    for (int tc = 0; tc < 4; ++tc) {
      const unsigned short* bp = &sB[(wn * 64 + tc * 16 + l15) * LDT + hi * 8];
      union { v16bf v; v8bf h[2]; } t;
      t.h[0] = *(const v8bf*)(bp);
      t.h[1] = *(const v8bf*)(bp + 16);
      bf[tc] = t.v;
    }

#pragma unroll
    for (int tr = 0; tr < 2; ++tr)
#pragma unroll
      for (int tc = 0; tc < 4; ++tc)
        acc[tr][tc] = __builtin_amdgcn_wmma_f32_16x16x32_bf16(
            false, af[tr], false, bf[tc], (short)0, acc[tr][tc], false, false);

    __syncthreads();
  }

  // epilogue: C 16x16 f32 layout (VGPR i: lanes0-15 M=i, lanes16-31 M=8+i)
#pragma unroll
  for (int tr = 0; tr < 2; ++tr) {
#pragma unroll
    for (int tc = 0; tc < 4; ++tc) {
      const long col  = blockN + wn * 64 + tc * 16 + l15;
      const long row0 = blockM + wm * 32 + tr * 16 + hi * 8;
      float vv[8];
#pragma unroll
      for (int i = 0; i < 8; ++i) {
        float v = acc[tr][tc][i];
        if (HAS_BIAS) v += bias[col];
        if (ACT == 1) v = fmaxf(v, 0.0f);
        vv[i] = v;
      }
      if (HAS_CF) {
        float* cf = Cf + row0 * (long)ldcf + col;
#pragma unroll
        for (int i = 0; i < 8; ++i) { *cf = vv[i]; cf += ldcf; }
      }
      if (HAS_CB) {
        if (CBT) {   // transposed: 8 consecutive rows -> one b128 store
          union { unsigned short h[8]; uint4 q; } pk;
#pragma unroll
          for (int i = 0; i < 8; ++i) pk.h[i] = f2bf_bits(vv[i]);
          *(uint4*)(Cb + col * (long)ldcb + row0) = pk.q;
        } else {
          unsigned short* cb = Cb + row0 * (long)ldcb + col;
#pragma unroll
          for (int i = 0; i < 8; ++i) { *cb = f2bf_bits(vv[i]); cb += ldcb; }
        }
      }
    }
  }
}

// ---------------- attention kernels -----------------------------------------

// H: [3*N, 128] pre-activation hidden (first 16 cols real); w2: [16].
// tanh applied here (branch-free), then dot + 3-way softmax -> beta [N,3].
extern "C" __global__ void __launch_bounds__(256)
k_attn_beta(const float* __restrict__ H, const float* __restrict__ w2,
            float* __restrict__ beta, int n_nodes) {
  int n = blockIdx.x * blockDim.x + threadIdx.x;
  if (n >= n_nodes) return;
  float s[3];
#pragma unroll
  for (int v = 0; v < 3; ++v) {
    const float* h = H + (((long)v * n_nodes + n) << 7);
    float a = 0.f;
#pragma unroll
    for (int j = 0; j < 16; ++j) {
      float x = h[j];
      float t = 1.0f - 2.0f / (__expf(2.0f * x) + 1.0f);   // tanh, branch-free
      a += t * w2[j];
    }
    s[v] = a;
  }
  float m = fmaxf(s[0], fmaxf(s[1], s[2]));
  float x0 = __expf(s[0] - m), x1 = __expf(s[1] - m), x2 = __expf(s[2] - m);
  float inv = 1.0f / (x0 + x1 + x2);
  beta[(long)n * 3 + 0] = x0 * inv;
  beta[(long)n * 3 + 1] = x1 * inv;
  beta[(long)n * 3 + 2] = x2 * inv;
}

// emb = sum_v beta[n,v] * e_v[n,:], d fixed at 256 (i>>8 = node index)
extern "C" __global__ void __launch_bounds__(256)
k_attn_emb(const float* __restrict__ e_base, const float* __restrict__ beta,
           float* __restrict__ emb, unsigned short* __restrict__ emb_bf,
           long n_nodes) {
  long total = n_nodes << 8;
  long i = (long)blockIdx.x * blockDim.x + threadIdx.x;
  if (i >= total) return;
  long n = i >> 8;
  float v = beta[n * 3 + 0] * e_base[i] +
            beta[n * 3 + 1] * e_base[total + i] +
            beta[n * 3 + 2] * e_base[2 * total + i];
  emb[i] = v;
  emb_bf[i] = f2bf_bits(v);
}

// ---------------- host orchestration ----------------------------------------

extern "C" void kernel_launch(void* const* d_in, const int* in_sizes, int n_in,
                              void* d_out, int out_size, void* d_ws, size_t ws_size,
                              hipStream_t stream) {
  (void)in_sizes; (void)n_in; (void)out_size; (void)ws_size;
  constexpr int  NN  = 8192;
  constexpr long NNl = 8192;

  const float* adj[3] = {(const float*)d_in[0], (const float*)d_in[1], (const float*)d_in[2]};
  const float* fea[3] = {(const float*)d_in[3], (const float*)d_in[4], (const float*)d_in[5]};
  const float* W1v[3] = {(const float*)d_in[6], (const float*)d_in[8], (const float*)d_in[10]};
  const float* W2v[3] = {(const float*)d_in[7], (const float*)d_in[9], (const float*)d_in[11]};
  const int    Dv[3]  = {512, 512, 1024};
  const float* att_W1 = (const float*)d_in[12];
  const float* att_b1 = (const float*)d_in[13];
  const float* att_W2 = (const float*)d_in[14];
  const float* enc1_W = (const float*)d_in[15]; const float* enc1_b = (const float*)d_in[16];
  const float* enc2_W = (const float*)d_in[17]; const float* enc2_b = (const float*)d_in[18];
  const float* enc3_W = (const float*)d_in[19]; const float* enc3_b = (const float*)d_in[20];
  const float* z_W    = (const float*)d_in[21]; const float* z_b    = (const float*)d_in[22];
  const float* dec1_W = (const float*)d_in[23]; const float* dec1_b = (const float*)d_in[24];
  const float* dec2_W = (const float*)d_in[25]; const float* dec2_b = (const float*)d_in[26];
  const float* dec3_W = (const float*)d_in[27]; const float* dec3_b = (const float*)d_in[28];
  const float* xbar_W = (const float*)d_in[29]; const float* xbar_b = (const float*)d_in[30];

  // ---- workspace carve ----
  char* p = (char*)d_ws;
  auto take = [&](size_t bytes) -> char* {
    char* r = p;
    p += (bytes + 255) & ~(size_t)255;
    return r;
  };
  unsigned short* adjB = (unsigned short*)take(NNl * NNl * 2);            // bf16 adj
  unsigned short* feaT = (unsigned short*)take((size_t)1024 * NNl * 2);   // X^T [D x N]
  unsigned short* axB  = (unsigned short*)take(NNl * 1024 * 2);           // AX  [N x D]
  unsigned short* zT   = (unsigned short*)take((size_t)512 * NNl * 2);    // Z^T [H x N]
  unsigned short* azB  = (unsigned short*)take(NNl * 512 * 2);            // AZ  [N x H]
  unsigned short* wT   = (unsigned short*)take((size_t)2048 * 2048 * 2);  // W^T padded
  unsigned short* eB   = (unsigned short*)take(3 * NNl * 256 * 2);        // [3N x 256]
  float*          hPad = (float*)take(3 * NNl * 128 * 4);                 // attn hidden
  float*          zPad = (float*)take(NNl * 128 * 4);                     // z padded
  float*          bPad = (float*)take(2048 * 4);                          // padded bias
  unsigned short* actA = (unsigned short*)take(NNl * 2048 * 2);
  unsigned short* actB = (unsigned short*)take(NNl * 2048 * 2);

  float* out      = (float*)d_out;
  float* e_out    = out;                       // e0,e1,ec
  float* emb_out  = out + 3 * NNl * 256;
  float* xbar_out = out + 4 * NNl * 256;
  float* z_out    = out + 5 * NNl * 256;
  float* beta_out = out + 5 * NNl * 256 + NNl * 10;

  auto cvt = [&](const float* s, unsigned short* d, long n) {
    long n8 = n / 8;
    long b = (n8 + 255) / 256; if (b > 16384) b = 16384;
    k_cvt_bf16<<<dim3((unsigned)b), dim3(256), 0, stream>>>(s, d, n8);
  };
  auto cvtT = [&](const float* s, unsigned short* d, int R, int C) {
    k_cvt_t<<<dim3(C / 32, R / 32), dim3(256), 0, stream>>>(s, d, R, C);
  };
  auto cvtW = [&](const float* s, int K, int Ncol, unsigned short* d, int kpShift, int Np) {
    long t = (long)Np << kpShift;
    long b = (t + 255) / 256; if (b > 4096) b = 4096;
    k_cvt_pad_t<<<dim3((unsigned)b), dim3(256), 0, stream>>>(s, K, Ncol, d, kpShift, Np);
  };
  auto padBias = [&](const float* s, int n, int np) {
    k_pad_bias<<<dim3((np + 255) / 256), dim3(256), 0, stream>>>(s, n, bPad, np);
  };
  auto grid = [&](int M, int Ncols) { return dim3(Ncols / BN, M / BM); };

  // ---- three GCN views: e_v = adj @ relu((adj @ X) @ W1) @ W2 ----
  for (int v = 0; v < 3; ++v) {
    const int D = Dv[v], H = D / 2;
    const int dShift = (D == 512) ? 9 : 10, hShift = dShift - 1;
    cvt(adj[v], adjB, NNl * NNl);
    cvtT(fea[v], feaT, NN, D);                         // X^T [D x N]
    cvtW(W1v[v], D, H, wT, dShift, H);                 // W1^T [H x D]
    // AX = adj @ X                    [N x D]  (bf16 out)
    k_gemm<0, false, false, true, false><<<grid(NN, D), dim3(256), 0, stream>>>(
        adjB, NN, feaT, NN, NN, nullptr, nullptr, 0, axB, D);
    // Z^T = relu(AX @ W1)^T           [H x N]  (transposed bf16 out)
    k_gemm<1, false, false, true, true><<<grid(NN, H), dim3(256), 0, stream>>>(
        axB, D, wT, D, D, nullptr, nullptr, 0, zT, NN);
    // AZ = adj @ Z                    [N x H]  (B = Z^T)
    k_gemm<0, false, false, true, false><<<grid(NN, H), dim3(256), 0, stream>>>(
        adjB, NN, zT, NN, NN, nullptr, nullptr, 0, azB, H);
    // e_v = AZ @ W2                   [N x 256] -> d_out f32 + eB bf16
    cvtW(W2v[v], H, 256, wT, hShift, 256);
    k_gemm<0, false, true, true, false><<<grid(NN, 256), dim3(256), 0, stream>>>(
        azB, H, wT, H, H, nullptr,
        e_out + (long)v * NN * 256, 256, eB + (long)v * NN * 256, 256);
  }

  // ---- attention over views ----
  cvtW(att_W1, 256, 16, wT, 8, 128);                   // pad 16 -> 128 cols
  padBias(att_b1, 16, 128);
  // hPad = [e0;e1;ec] @ att_W1 + b1   [3N x 128] (tanh applied in k_attn_beta)
  k_gemm<0, true, true, false, false><<<grid(3 * NN, 128), dim3(256), 0, stream>>>(
      eB, 256, wT, 256, 256, bPad, hPad, 128, nullptr, 0);
  k_attn_beta<<<dim3((NN + 255) / 256), dim3(256), 0, stream>>>(hPad, att_W2, beta_out, NN);
  k_attn_emb<<<dim3((unsigned)((NNl * 256 + 255) / 256)), dim3(256), 0, stream>>>(
      e_out, beta_out, emb_out, actA, NNl);            // emb bf16 -> actA [N x 256]

  // ---- autoencoder (padded: 500->512, 2000->2048, 10->128) ----
  cvtW(enc1_W, 256, 500, wT, 8, 512);
  padBias(enc1_b, 500, 512);
  k_gemm<1, true, false, true, false><<<grid(NN, 512), dim3(256), 0, stream>>>(
      actA, 256, wT, 256, 256, bPad, nullptr, 0, actB, 512);
  cvtW(enc2_W, 500, 500, wT, 9, 512);
  padBias(enc2_b, 500, 512);
  k_gemm<1, true, false, true, false><<<grid(NN, 512), dim3(256), 0, stream>>>(
      actB, 512, wT, 512, 512, bPad, nullptr, 0, actA, 512);
  cvtW(enc3_W, 500, 2000, wT, 9, 2048);
  padBias(enc3_b, 2000, 2048);
  k_gemm<1, true, false, true, false><<<grid(NN, 2048), dim3(256), 0, stream>>>(
      actA, 512, wT, 512, 512, bPad, nullptr, 0, actB, 2048);
  cvtW(z_W, 2000, 10, wT, 11, 128);
  padBias(z_b, 10, 128);
  // z (padded) -> zPad f32 + actA bf16
  k_gemm<0, true, true, true, false><<<grid(NN, 128), dim3(256), 0, stream>>>(
      actB, 2048, wT, 2048, 2048, bPad, zPad, 128, actA, 128);
  k_copy_z<<<dim3((unsigned)((NNl * 16 + 255) / 256)), dim3(256), 0, stream>>>(
      zPad, z_out, NNl * 16);
  cvtW(dec1_W, 10, 2000, wT, 7, 2048);
  padBias(dec1_b, 2000, 2048);
  k_gemm<1, true, false, true, false><<<grid(NN, 2048), dim3(256), 0, stream>>>(
      actA, 128, wT, 128, 128, bPad, nullptr, 0, actB, 2048);
  cvtW(dec2_W, 2000, 500, wT, 11, 512);
  padBias(dec2_b, 500, 512);
  k_gemm<1, true, false, true, false><<<grid(NN, 512), dim3(256), 0, stream>>>(
      actB, 2048, wT, 2048, 2048, bPad, nullptr, 0, actA, 512);
  cvtW(dec3_W, 500, 500, wT, 9, 512);
  padBias(dec3_b, 500, 512);
  k_gemm<1, true, false, true, false><<<grid(NN, 512), dim3(256), 0, stream>>>(
      actA, 512, wT, 512, 512, bPad, nullptr, 0, actB, 512);
  cvtW(xbar_W, 500, 256, wT, 9, 256);
  // xbar_b is exactly 256 wide: use directly
  k_gemm<0, true, true, false, false><<<grid(NN, 256), dim3(256), 0, stream>>>(
      actB, 512, wT, 512, 512, xbar_b, xbar_out, 256, nullptr, 0);
}
